// GNCC_19404662243709
// MI455X (gfx1250) — matrix operand order repaired and checked by reading, not compiled, not run on hardware
//
#include <hip/hip_runtime.h>
#include <hip/hip_bf16.h>
#include <stdint.h>

// ---------------------------------------------------------------------------
// NNConv (edge-conditioned GNN) for MI455X / gfx1250.
// Dominant work: per-edge MLP  relu(attr@w1+b1)@w2+b2  ->  426 GFLOP of GEMM.
// Strategy: f16 WMMA (v_wmma_f32_16x16x32_f16) with f32 accumulate, fully
// fused so the [E,1024] per-edge weight tensor never reaches HBM.
// ---------------------------------------------------------------------------

typedef _Float16 v8h  __attribute__((ext_vector_type(8)));
typedef _Float16 v16h __attribute__((ext_vector_type(16)));
typedef float    v8f  __attribute__((ext_vector_type(8)));

union H16 { v16h v; v8h h[2]; };

#define CH    32      // IN_CH == HID_CH
#define EDIM  16
#define KTOT  1024    // CH*CH (edge-MLP hidden / output width)
#define TE    16      // edges per workgroup (one WMMA M-tile)

// ---------------------------------------------------------------------------
// Weight conversion: w2 [1024,1024] f32 (k-major) -> w2t [n][k] f16.
// B-fragment per lane is then 16 contiguous f16 (32B) at w2t[n*1024 + kbase].
// ---------------------------------------------------------------------------
__global__ void GNCC_conv_w2(const float* __restrict__ w2, _Float16* __restrict__ w2t) {
    int idx = blockIdx.x * blockDim.x + threadIdx.x;
    if (idx >= KTOT * KTOT) return;
    int n = idx >> 10, k = idx & 1023;
    w2t[idx] = (_Float16)w2[k * KTOT + n];
}

// w1 [16,1024] f32 -> w1t [n][32] f16 with K padded 16->32 (zeros), so GEMM1
// uses the same 16x16x32 WMMA shape with the upper half of K contributing 0.
__global__ void GNCC_conv_w1(const float* __restrict__ w1, _Float16* __restrict__ w1t) {
    int idx = blockIdx.x * blockDim.x + threadIdx.x;
    if (idx >= KTOT * 32) return;
    int n = idx >> 5, k = idx & 31;
    w1t[idx] = (k < EDIM) ? (_Float16)w1[k * KTOT + n] : (_Float16)0.0f;
}

// ---------------------------------------------------------------------------
// Fused edge kernel: one workgroup = 16 edges, 8 waves (wave32).
//   1) t = relu(attr@w1 + b1)          (1 WMMA per 16-col tile, K zero-padded)
//      t [16 x 1024] f16 staged in LDS (shared A-matrix for GEMM2)
//   2) h = t@w2 + b2                   (per wave: 128 cols, 256 WMMAs)
//   3) msg[e][o] = sum_i x[src[e]][i] * h[e][i*32+o]   folded into epilogue
//      via LDS f32 atomics (ds_add_f32), then scattered with global atomics.
// ---------------------------------------------------------------------------
__global__ __launch_bounds__(256)
void GNCC_edge_fused(const float* __restrict__ xin,
                     const int*   __restrict__ src,
                     const int*   __restrict__ dst,
                     const float* __restrict__ eattr,
                     const _Float16* __restrict__ w1t,  // [1024][32] f16
                     const float* __restrict__ b1,      // [1024]
                     const _Float16* __restrict__ w2t,  // [1024][1024] f16 n-major
                     const float* __restrict__ b2,      // [1024]
                     float* __restrict__ agg,           // [N][32]
                     float* __restrict__ cnt,           // [N]
                     int E)
{
    __shared__ _Float16 lds_t[TE * KTOT];      // 32 KB  (GEMM2 A-matrix)
    __shared__ _Float16 lds_attr[TE * 32];     // 1 KB   (GEMM1 A, K padded)
    __shared__ float    lds_xs[TE * CH];       // gathered x[src]
    __shared__ float    lds_msg[TE * CH];      // per-edge message accumulator

    const int tid  = threadIdx.x;
    const int wave = tid >> 5;                 // 0..7
    const int lane = tid & 31;
    const int hi   = lane >> 4;                // lane half (WMMA layout)
    const int l16  = lane & 15;
    const int e0   = blockIdx.x * TE;

    // ---- stage: gather x[src], edge_attr (f16, K-padded), zero msg ----
    for (int idx = tid; idx < TE * 32; idx += 256) {
        int e = idx >> 5, c = idx & 31;
        int eg = e0 + e;
        float xv = 0.0f;
        if (eg < E) xv = xin[(size_t)src[eg] * CH + c];
        lds_xs[idx]  = xv;
        lds_msg[idx] = 0.0f;
        _Float16 av = (_Float16)0.0f;
        if (c < EDIM && eg < E) av = (_Float16)eattr[(size_t)eg * EDIM + c];
        lds_attr[idx] = av;                    // cols 16..31 are zero padding
    }
    __syncthreads();

    // ---- GEMM1: t = relu(attr @ w1 + b1), write f16 into LDS ----
    // K order used here for A and B: lane<16 holds K {0..15}, lane>=16 {16..31}
    H16 a1;
    a1.v = *(const v16h*)&lds_attr[l16 * 32 + hi * 16];
    #pragma unroll
    for (int nt = 0; nt < 8; ++nt) {
        int n = wave * 128 + nt * 16 + l16;
        const v16h bfrag = *(const v16h*)&w1t[(size_t)n * 32 + hi * 16];
        v8f c = {};
        c = __builtin_amdgcn_wmma_f32_16x16x32_f16(false, a1.v, false, bfrag,
                                                   (short)0, c, false, false);
        float bb = b1[n];
        #pragma unroll
        for (int r = 0; r < 8; ++r) {          // C layout: M = r + 8*hi, N = l16
            int m = r + hi * 8;
            float v = c[r] + bb;
            v = v > 0.0f ? v : 0.0f;
            lds_t[m * KTOT + n] = (_Float16)v;
        }
    }
    __syncthreads();

    // ---- GEMM2: acc[16 x 128] = t @ w2 (this wave's 128 columns) ----
    v8f acc[8];
    #pragma unroll
    for (int i = 0; i < 8; ++i) acc[i] = (v8f){};
    const int n_base = wave * 128;
    for (int k0 = 0; k0 < KTOT; k0 += 32) {
        H16 a;
        a.v = *(const v16h*)&lds_t[l16 * KTOT + k0 + hi * 16];
        #pragma unroll
        for (int nt = 0; nt < 8; ++nt) {
            int n = n_base + nt * 16 + l16;
            const v16h b = *(const v16h*)&w2t[(size_t)n * KTOT + k0 + hi * 16];
            acc[nt] = __builtin_amdgcn_wmma_f32_16x16x32_f16(false, a.v, false, b,
                                                             (short)0, acc[nt],
                                                             false, false);
        }
    }

    // ---- epilogue: fold bias + per-edge matvec message into LDS ----
    // column c = i*32 + o  ->  msg[e][o] += xs[e][i] * (acc + b2[c])
    #pragma unroll
    for (int nt = 0; nt < 8; ++nt) {
        int c = n_base + nt * 16 + l16;
        int i = c >> 5, o = c & 31;
        float bb = b2[c];
        #pragma unroll
        for (int r = 0; r < 8; ++r) {
            int m = r + hi * 8;
            float hval = acc[nt][r] + bb;
            atomicAdd(&lds_msg[m * CH + o], lds_xs[m * CH + i] * hval);
        }
    }
    __syncthreads();

    // ---- scatter to destination nodes ----
    for (int idx = tid; idx < TE * CH; idx += 256) {
        int e = idx >> 5, o = idx & 31;
        int eg = e0 + e;
        if (eg < E) atomicAdd(&agg[(size_t)dst[eg] * CH + o], lds_msg[idx]);
    }
    if (tid < TE) {
        int eg = e0 + tid;
        if (eg < E) atomicAdd(&cnt[dst[eg]], 1.0f);
    }
}

// ---------------------------------------------------------------------------
// Node update: h = relu(agg/max(cnt,1) + x@root + bias)   (41 MFLOP -> VALU)
// ---------------------------------------------------------------------------
__global__ void GNCC_node_update(const float* __restrict__ xin,
                                 const float* __restrict__ agg,
                                 const float* __restrict__ cnt,
                                 const float* __restrict__ root,  // [32][32]
                                 const float* __restrict__ bias,  // [32]
                                 float* __restrict__ hout, int N)
{
    int idx = blockIdx.x * blockDim.x + threadIdx.x;
    if (idx >= N * CH) return;
    int n = idx >> 5, o = idx & 31;
    float c = cnt[n];
    c = c > 1.0f ? c : 1.0f;
    float v = agg[idx] / c + bias[o];
    const float* xr = xin + (size_t)n * CH;
    #pragma unroll
    for (int i = 0; i < CH; ++i) v += xr[i] * root[i * CH + o];
    hout[idx] = v > 0.0f ? v : 0.0f;
}

// Final classifier: out = h2 @ lin_w + lin_b   (32 -> 10, negligible)
__global__ void GNCC_final_linear(const float* __restrict__ h,
                                  const float* __restrict__ lw,   // [32][10]
                                  const float* __restrict__ lb,   // [10]
                                  float* __restrict__ out, int N)
{
    int idx = blockIdx.x * blockDim.x + threadIdx.x;
    if (idx >= N * 10) return;
    int n = idx / 10, c = idx % 10;
    float acc = lb[c];
    #pragma unroll
    for (int i = 0; i < CH; ++i) acc += h[(size_t)n * CH + i] * lw[i * 10 + c];
    out[idx] = acc;
}

static inline size_t align256(size_t x) { return (x + 255) & ~(size_t)255; }

extern "C" void kernel_launch(void* const* d_in, const int* in_sizes, int n_in,
                              void* d_out, int out_size, void* d_ws, size_t ws_size,
                              hipStream_t stream) {
    const float* x      = (const float*)d_in[0];
    const int*   ei     = (const int*)  d_in[1];
    const float* eattr  = (const float*)d_in[2];
    const float* e1_w1  = (const float*)d_in[3];
    const float* e1_b1  = (const float*)d_in[4];
    const float* e1_w2  = (const float*)d_in[5];
    const float* e1_b2  = (const float*)d_in[6];
    const float* root1  = (const float*)d_in[7];
    const float* bias1  = (const float*)d_in[8];
    const float* e2_w1  = (const float*)d_in[9];
    const float* e2_b1  = (const float*)d_in[10];
    const float* e2_w2  = (const float*)d_in[11];
    const float* e2_b2  = (const float*)d_in[12];
    const float* root2  = (const float*)d_in[13];
    const float* bias2  = (const float*)d_in[14];
    const float* lin_w  = (const float*)d_in[15];
    const float* lin_b  = (const float*)d_in[16];

    const int N = in_sizes[0] / CH;
    const int E = in_sizes[1] / 2;
    const int* src = ei;
    const int* dst = ei + E;

    // workspace partition (aligned)
    char* p = (char*)d_ws;
    _Float16* w1t1 = (_Float16*)p; p += align256((size_t)KTOT * 32 * 2);
    _Float16* w2t1 = (_Float16*)p; p += align256((size_t)KTOT * KTOT * 2);
    _Float16* w1t2 = (_Float16*)p; p += align256((size_t)KTOT * 32 * 2);
    _Float16* w2t2 = (_Float16*)p; p += align256((size_t)KTOT * KTOT * 2);
    float* agg = (float*)p;        p += align256((size_t)N * CH * 4);
    float* cnt = (float*)p;        p += align256((size_t)N * 4);
    float* h1  = (float*)p;        p += align256((size_t)N * CH * 4);
    float* h2  = (float*)p;        p += align256((size_t)N * CH * 4);

    const int TPB = 256;
    // weight conversion (f32 -> f16, transposed / K-padded)
    GNCC_conv_w1<<<(KTOT * 32 + TPB - 1) / TPB, TPB, 0, stream>>>(e1_w1, w1t1);
    GNCC_conv_w2<<<(KTOT * KTOT + TPB - 1) / TPB, TPB, 0, stream>>>(e1_w2, w2t1);
    GNCC_conv_w1<<<(KTOT * 32 + TPB - 1) / TPB, TPB, 0, stream>>>(e2_w1, w1t2);
    GNCC_conv_w2<<<(KTOT * KTOT + TPB - 1) / TPB, TPB, 0, stream>>>(e2_w2, w2t2);

    const int edge_blocks = (E + TE - 1) / TE;
    const int node_blocks = (N * CH + TPB - 1) / TPB;

    // ---- layer 1 ----
    hipMemsetAsync(agg, 0, (size_t)N * CH * 4, stream);
    hipMemsetAsync(cnt, 0, (size_t)N * 4, stream);
    GNCC_edge_fused<<<edge_blocks, TPB, 0, stream>>>(x, src, dst, eattr,
                                                     w1t1, e1_b1, w2t1, e1_b2,
                                                     agg, cnt, E);
    GNCC_node_update<<<node_blocks, TPB, 0, stream>>>(x, agg, cnt, root1, bias1, h1, N);

    // ---- layer 2 ----
    hipMemsetAsync(agg, 0, (size_t)N * CH * 4, stream);
    hipMemsetAsync(cnt, 0, (size_t)N * 4, stream);
    GNCC_edge_fused<<<edge_blocks, TPB, 0, stream>>>(h1, src, dst, eattr,
                                                     w1t2, e2_b1, w2t2, e2_b2,
                                                     agg, cnt, E);
    GNCC_node_update<<<node_blocks, TPB, 0, stream>>>(h1, agg, cnt, root2, bias2, h2, N);

    // ---- classifier ----
    GNCC_final_linear<<<(N * 10 + TPB - 1) / TPB, TPB, 0, stream>>>(h2, lin_w, lin_b,
                                                                    (float*)d_out, N);
}
